// Big_Bird_Attention_57964878627285
// MI455X (gfx1250) — matrix operand order repaired and checked by reading, not compiled, not run on hardware
//
#include <hip/hip_runtime.h>

// ---------------------------------------------------------------------------
// BigBird block (B=2, S=4096, DM=1024, H=16, DH=64, BLK=64, M=64, R=3, G=8)
// All matmuls on v_wmma_f32_16x16x32_f16 (wave32).
// Weights are pre-swizzled into WMMA B-fragment order so GEMM B fragments are
// single contiguous 32B/lane global loads (no LDS staging, no packing VALU).
// Workspace layout (requires ws_size >= 140 MB):
//   [0,16MB)    x_f16  -> later reused as ctx_f16
//   [16,28MB)   6 weight matrices f16, fragment-swizzled (2MB each)
//   [28,44MB)   q_f16  -> later reused as inner_f16 (FFN hidden)
//   [44,60MB)   k_f16  -> later reused as h_f16
//   [60,76MB)   v_f16
//   [76,108MB)  f32 buf (attn_out, later ffn out2)
//   [108,140MB) h_f32
// ---------------------------------------------------------------------------

typedef __attribute__((ext_vector_type(16))) _Float16 v16h;
typedef __attribute__((ext_vector_type(8)))  float    v8f;

#define S_LEN   4096
#define DM_DIM  1024
#define H_NUM   16
#define M_BLKS  64
#define NEG_INF (-1.0e9f)

__device__ __forceinline__ v8f wmma16(v16h a, v16h b, v8f c) {
  return __builtin_amdgcn_wmma_f32_16x16x32_f16(false, a, false, b, (short)0, c,
                                                false, false);
}

// A-fragment (16x32 f16, M x K), tile row-major [M][K].
// lanes 0-15 -> K {0..15}; lanes 16-31 -> K {16..31};
// row m = lane%16.  Both 8-element runs are contiguous -> 2x ds_load_b128.
__device__ __forceinline__ v16h frag_a(const _Float16* t, int ld) {
  const int lane = threadIdx.x & 31;
  const int m  = lane & 15;
  const int kb = (lane < 16) ? 0 : 16;
  v16h a;
#pragma unroll
  for (int j = 0; j < 8; ++j) a[j]     = t[m * ld + kb + j];
#pragma unroll
  for (int j = 0; j < 8; ++j) a[8 + j] = t[m * ld + 8 + kb + j];
  return a;
}

// B-fragment (32x16 f16, K x N) from a [N][K]-layout tile:
// B[k][n] = t[n*ld + k];  lane n = lane%16, K = (lane<16?0:16)+j.
// Contiguous in j -> 2x ds_load_b128.
__device__ __forceinline__ v16h frag_b_tr(const _Float16* t, int ld) {
  const int lane = threadIdx.x & 31;
  const int n  = lane & 15;
  const int k0 = (lane < 16) ? 0 : 16;
  v16h b;
#pragma unroll
  for (int j = 0; j < 16; ++j) b[j] = t[n * ld + k0 + j];
  return b;
}

// ---------------------------------------------------------------------------
// fp32 -> f16 elementwise conversion (activations)
// ---------------------------------------------------------------------------
__global__ __launch_bounds__(256) void cvt_f32_f16(const float* __restrict__ s,
                                                   _Float16* __restrict__ d,
                                                   int n) {
  int i = blockIdx.x * 256 + threadIdx.x;
  if (i < n) d[i] = (_Float16)s[i];
}

// ---------------------------------------------------------------------------
// fp32 -> f16 weight conversion with WMMA-B-fragment swizzle.
// Input W[K=1024][N=1024] row-major.  Output: fragments of 32(K) x 16(N),
// fragment (kt, nt) stored at ((kt*64 + nt)*512), per-lane 16 contiguous f16:
//   out[frag*512 + lane*16 + j] = W[kt*32 + (lane<16?0:16) + j][nt*16 + lane%16]
// ---------------------------------------------------------------------------
__global__ __launch_bounds__(256)
void cvt_w_swizzle(const float* __restrict__ w, _Float16* __restrict__ o) {
  int t    = blockIdx.x * 256 + threadIdx.x;
  int lane = t & 31;
  int nt   = (t >> 5) & 63;
  int kt   = t >> 11;  // 0..31
  int n    = lane & 15;
  int k0   = kt * 32 + ((lane < 16) ? 0 : 16);
  _Float16*    dst = o + ((size_t)(kt * 64 + nt)) * 512 + lane * 16;
  const float* src = w + (size_t)k0 * DM_DIM + nt * 16 + n;
#pragma unroll
  for (int j = 0; j < 16; ++j) dst[j] = (_Float16)src[(size_t)j * DM_DIM];
}

// ---------------------------------------------------------------------------
// Tiled WMMA GEMM: C[M,N] = A[M,K] * B[K,N] (+bias) (optional ReLU).
// Block tile 128x256, BK=32, 256 thr = 8 waves; wave (wm 0..1, wn 0..3) owns
// a 64x64 sub-tile = 4x4 WMMA tiles (16 accumulators).
// Per k-step / wave: 16 WMMA vs 8 ds_load_b128 (A) + 8 global b128 (B).
// A staged via LDS; B fragments straight from L2-resident swizzled weights.
// Requires M%128==0, N%256==0, K%32==0.
// ---------------------------------------------------------------------------
template <bool RELU, bool OUT_F32>
__global__ __launch_bounds__(256)
void gemm_wmma_f16(const _Float16* __restrict__ A,
                   const _Float16* __restrict__ Bsw,
                   const float* __restrict__ bias,
                   float* __restrict__ outF, _Float16* __restrict__ outH,
                   int Mdim, int Ndim, int Kdim) {
  __shared__ _Float16 As[128 * 32];
  (void)Mdim;

  const int tid  = threadIdx.x;
  const int lane = tid & 31;
  const int w    = tid >> 5;
  const int wm   = w >> 2;  // 0..1 : 64 rows
  const int wn   = w & 3;   // 0..3 : 64 cols
  const int bn0  = blockIdx.x * 256;
  const int bm0  = blockIdx.y * 128;
  const int ntiles = Ndim >> 4;

  v8f acc[4][4];
  const v8f vz = {0.f, 0.f, 0.f, 0.f, 0.f, 0.f, 0.f, 0.f};
#pragma unroll
  for (int i = 0; i < 4; ++i)
#pragma unroll
    for (int j = 0; j < 4; ++j) acc[i][j] = vz;

  for (int k0 = 0; k0 < Kdim; k0 += 32) {
    {  // stage A tile 128x32
      int r = tid >> 1, cs = (tid & 1) * 16;
      const float4* sp =
          (const float4*)(A + (size_t)(bm0 + r) * Kdim + k0 + cs);
      float4* dp = (float4*)&As[r * 32 + cs];
      dp[0] = sp[0];
      dp[1] = sp[1];
      if (k0 + 32 < Kdim)  // global_prefetch_b8
        __builtin_prefetch(A + (size_t)(bm0 + r) * Kdim + k0 + 32 + cs, 0, 1);
    }
    __syncthreads();

    // B fragments straight from swizzled global weights
    const int kt = k0 >> 5;
    const _Float16* bbase =
        Bsw + ((size_t)kt * ntiles + (bn0 >> 4) + wn * 4) * 512 + lane * 16;
    v16h bf[4];
#pragma unroll
    for (int nt = 0; nt < 4; ++nt) bf[nt] = *(const v16h*)(bbase + nt * 512);

#pragma unroll
    for (int mt = 0; mt < 4; ++mt) {
      v16h a = frag_a(&As[(wm * 64 + mt * 16) * 32], 32);
#pragma unroll
      for (int nt = 0; nt < 4; ++nt)
        acc[mt][nt] = wmma16(a, bf[nt], acc[mt][nt]);
    }
    __syncthreads();
  }

  // Epilogue: C-layout (lane n = lane%16, reg r -> m = r + (lane<16?0:8))
  const int n  = lane & 15;
  const int mb = (lane < 16) ? 0 : 8;
#pragma unroll
  for (int mt = 0; mt < 4; ++mt) {
#pragma unroll
    for (int nt = 0; nt < 4; ++nt) {
      int gcol = bn0 + wn * 64 + nt * 16 + n;
      float bv = bias ? bias[gcol] : 0.f;
#pragma unroll
      for (int r = 0; r < 8; ++r) {
        int grow = bm0 + wm * 64 + mt * 16 + mb + r;
        float v  = acc[mt][nt][r] + bv;
        if (RELU) v = fmaxf(v, 0.f);
        size_t o = (size_t)grow * Ndim + gcol;
        if (OUT_F32) outF[o] = v;
        else         outH[o] = (_Float16)v;
      }
    }
  }
}

// ---------------------------------------------------------------------------
// BigBird flash attention. One workgroup (256 thr = 8 waves) per (b,h,m).
// The key-block schedule (block ids + dedup/keep flags) is materialized into
// LDS tables up front so the hot loop is branch-free and uniform:
//   interior m : 8 gathered blocks (window 3 + global 2 + random 3), keep mask
//   edge m     : all 64 blocks, keep = 1 (full attention, matches reference)
// Online softmax; QK^T and P*V on WMMA; V staged transposed for contiguous
// B-fragment ds_load_b128.
// ---------------------------------------------------------------------------
__global__ __launch_bounds__(256)
void bigbird_attn(const _Float16* __restrict__ q, const _Float16* __restrict__ k,
                  const _Float16* __restrict__ v, const float* __restrict__ mask,
                  const int* __restrict__ rand_blocks,
                  _Float16* __restrict__ ctx) {
  __shared__ _Float16 Qs[64 * 64];   // [q][dh]
  __shared__ _Float16 Ks[64 * 64];   // [key][dh]
  __shared__ _Float16 Vt[64 * 64];   // transposed: [dh][key]
  __shared__ _Float16 Ps[64 * 64];   // [q][key]
  __shared__ float    Ss[64 * 64];
  __shared__ float    mrow[64], lrow[64], corr[64], maskv[64];
  __shared__ int      idxs[64];
  __shared__ float    keepf[64];

  const int wg  = blockIdx.x;  // b*H*M + h*M + m
  const int m   = wg % M_BLKS;
  const int h   = (wg / M_BLKS) % H_NUM;
  const int b   = wg / (M_BLKS * H_NUM);
  const int tid = threadIdx.x;
  const int lane = tid & 31;
  const int w    = tid >> 5;
  const int qg   = w & 3;   // query row group (16 rows)
  const int dg   = w >> 2;  // dh half (32 cols)

  const bool edge = (m == 0) || (m == M_BLKS - 1);
  const int  nkb  = edge ? M_BLKS : 8;

  if (tid < 64) { mrow[tid] = -1e30f; lrow[tid] = 0.f; }
  if (edge) {
    if (tid < 64) { idxs[tid] = tid; keepf[tid] = 1.f; }
  } else if (tid == 0) {
    idxs[0] = (m - 1 + M_BLKS) % M_BLKS;
    idxs[1] = m;
    idxs[2] = (m + 1) % M_BLKS;
    idxs[3] = 0;
    idxs[4] = M_BLKS - 1;
    idxs[5] = rand_blocks[m * 3 + 0];
    idxs[6] = rand_blocks[m * 3 + 1];
    idxs[7] = rand_blocks[m * 3 + 2];
    for (int j = 0; j < 8; ++j) {
      float kp = 1.f;
      for (int i = 0; i < j; ++i)
        if (idxs[i] == idxs[j]) kp = 0.f;
      keepf[j] = kp;
    }
  }
  {  // stage Q tile (64 x 64)
    int r = tid >> 2, cs = (tid & 3) * 16;
    const float4* sp = (const float4*)(q + ((size_t)b * S_LEN + m * 64 + r) *
                                               DM_DIM + h * 64 + cs);
    float4* dp = (float4*)&Qs[r * 64 + cs];
    dp[0] = sp[0];
    dp[1] = sp[1];
  }
  __syncthreads();

  const float scale = 0.125f;  // 1/sqrt(64)
  const v8f vz = {0.f, 0.f, 0.f, 0.f, 0.f, 0.f, 0.f, 0.f};
  v8f c0 = vz, c1 = vz;

  for (int kb = 0; kb < nkb; ++kb) {
    const int   blkid = idxs[kb];
    const float kflag = keepf[kb];
    {  // stage K (row-major) and V (transposed) blocks
      int r = tid >> 2, cs = (tid & 3) * 16;
      size_t base = ((size_t)b * S_LEN + blkid * 64 + r) * DM_DIM + h * 64 + cs;
      ((float4*)&Ks[r * 64 + cs])[0] = ((const float4*)(k + base))[0];
      ((float4*)&Ks[r * 64 + cs])[1] = ((const float4*)(k + base))[1];
      float4 v0 = ((const float4*)(v + base))[0];
      float4 v1 = ((const float4*)(v + base))[1];
      const _Float16* ve0 = (const _Float16*)&v0;
      const _Float16* ve1 = (const _Float16*)&v1;
#pragma unroll
      for (int j = 0; j < 8; ++j) Vt[(cs + j) * 64 + r]     = ve0[j];
#pragma unroll
      for (int j = 0; j < 8; ++j) Vt[(cs + 8 + j) * 64 + r] = ve1[j];
    }
    if (tid < 64) maskv[tid] = mask[(size_t)b * S_LEN + blkid * 64 + tid] * kflag;
    __syncthreads();

    // ---- S = Q * K^T : wave covers (qg, key tiles {2*dg, 2*dg+1}) ----
#pragma unroll
    for (int kt2 = 0; kt2 < 2; ++kt2) {
      int kt = dg * 2 + kt2;
      v8f sc = vz;
#pragma unroll
      for (int ks = 0; ks < 2; ++ks) {
        v16h a  = frag_a(&Qs[(qg * 16) * 64 + ks * 32], 64);
        v16h bf = frag_b_tr(&Ks[(kt * 16) * 64 + ks * 32], 64);
        sc = wmma16(a, bf, sc);
      }
      int n = lane & 15, mb2 = (lane < 16) ? 0 : 8;
#pragma unroll
      for (int r = 0; r < 8; ++r)
        Ss[(qg * 16 + mb2 + r) * 64 + kt * 16 + n] = sc[r];
    }
    __syncthreads();

    // ---- online softmax: 4 threads per query row ----
    {
      int row = tid >> 2, seg = tid & 3, cbase = seg * 16;
      float sv[16];
      float mx = -3.0e38f;
#pragma unroll
      for (int j = 0; j < 16; ++j) {
        float s = Ss[row * 64 + cbase + j] * scale;
        if (maskv[cbase + j] <= 0.f) s = NEG_INF;
        sv[j] = s;
        mx = fmaxf(mx, s);
      }
      mx = fmaxf(mx, __shfl_xor(mx, 1));
      mx = fmaxf(mx, __shfl_xor(mx, 2));
      float mold = mrow[row];
      float mnew = fmaxf(mold, mx);
      float sum = 0.f;
#pragma unroll
      for (int j = 0; j < 16; ++j) {
        float p = __expf(sv[j] - mnew);
        Ps[row * 64 + cbase + j] = (_Float16)p;
        sum += p;
      }
      sum += __shfl_xor(sum, 1);
      sum += __shfl_xor(sum, 2);
      if (seg == 0) {
        float cr = __expf(mold - mnew);
        corr[row] = cr;
        lrow[row] = lrow[row] * cr + sum;
        mrow[row] = mnew;
      }
    }
    __syncthreads();

    // ---- rescale accumulators, then ctx += P * V ----
    {
      int mb2 = (lane < 16) ? 0 : 8;
#pragma unroll
      for (int r = 0; r < 8; ++r) {
        float cr = corr[qg * 16 + mb2 + r];
        c0[r] *= cr;
        c1[r] *= cr;
      }
#pragma unroll
      for (int kt = 0; kt < 2; ++kt) {  // key chunks of 32
        v16h a  = frag_a(&Ps[(qg * 16) * 64 + kt * 32], 64);
        v16h b0 = frag_b_tr(&Vt[(dg * 32 + 0) * 64 + kt * 32], 64);
        c0 = wmma16(a, b0, c0);
        v16h b1 = frag_b_tr(&Vt[(dg * 32 + 16) * 64 + kt * 32], 64);
        c1 = wmma16(a, b1, c1);
      }
    }
    __syncthreads();
  }

  {  // normalize and write ctx (f16, [b][s][h*64+dh])
    int n = lane & 15, mb2 = (lane < 16) ? 0 : 8;
#pragma unroll
    for (int r = 0; r < 8; ++r) {
      int row   = qg * 16 + mb2 + r;
      float inv = 1.f / lrow[row];
      size_t base =
          ((size_t)b * S_LEN + m * 64 + row) * DM_DIM + h * 64 + dg * 32;
      ctx[base + n]      = (_Float16)(c0[r] * inv);
      ctx[base + 16 + n] = (_Float16)(c1[r] * inv);
    }
  }
}

// ---------------------------------------------------------------------------
// y = LayerNorm(a + b) * g + beta ; one block per row (DM=1024, 256 threads).
// Optionally also writes the f16 copy used by the following GEMM.
// ---------------------------------------------------------------------------
template <bool W16>
__global__ __launch_bounds__(256)
void ln_res_kernel(const float* __restrict__ a, const float* __restrict__ b,
                   const float* __restrict__ g, const float* __restrict__ be,
                   float* __restrict__ outF, _Float16* __restrict__ outH) {
  __shared__ float red[8];
  const int row = blockIdx.x;
  const int t   = threadIdx.x;
  const size_t base = (size_t)row * DM_DIM;

  float vals[4];
  float s = 0.f;
#pragma unroll
  for (int i = 0; i < 4; ++i) {
    int c = t + i * 256;
    float x = a[base + c] + b[base + c];
    vals[i] = x;
    s += x;
  }
  for (int o = 16; o > 0; o >>= 1) s += __shfl_xor(s, o);
  if ((t & 31) == 0) red[t >> 5] = s;
  __syncthreads();
  if (t == 0) {
    float tt = 0.f;
    for (int i = 0; i < 8; ++i) tt += red[i];
    red[0] = tt;
  }
  __syncthreads();
  const float mu = red[0] * (1.0f / DM_DIM);
  __syncthreads();

  float s2 = 0.f;
#pragma unroll
  for (int i = 0; i < 4; ++i) {
    float d = vals[i] - mu;
    s2 += d * d;
  }
  for (int o = 16; o > 0; o >>= 1) s2 += __shfl_xor(s2, o);
  if ((t & 31) == 0) red[t >> 5] = s2;
  __syncthreads();
  if (t == 0) {
    float tt = 0.f;
    for (int i = 0; i < 8; ++i) tt += red[i];
    red[0] = tt;
  }
  __syncthreads();
  const float inv = rsqrtf(red[0] * (1.0f / DM_DIM) + 1e-12f);

#pragma unroll
  for (int i = 0; i < 4; ++i) {
    int c = t + i * 256;
    float y = (vals[i] - mu) * inv * g[c] + be[c];
    outF[base + c] = y;
    if (W16) outH[base + c] = (_Float16)y;
  }
}

// ---------------------------------------------------------------------------
// Host-side orchestration
// ---------------------------------------------------------------------------
extern "C" void kernel_launch(void* const* d_in, const int* in_sizes, int n_in,
                              void* d_out, int out_size, void* d_ws, size_t ws_size,
                              hipStream_t stream) {
  (void)in_sizes; (void)n_in; (void)out_size; (void)ws_size;

  const float* x     = (const float*)d_in[0];
  const float* mask  = (const float*)d_in[1];
  const int*   randb = (const int*)d_in[2];
  const float* Wq = (const float*)d_in[3];
  const float* bq = (const float*)d_in[4];
  const float* Wk = (const float*)d_in[5];
  const float* bk = (const float*)d_in[6];
  const float* Wv = (const float*)d_in[7];
  const float* bv = (const float*)d_in[8];
  const float* Wo = (const float*)d_in[9];
  const float* bo = (const float*)d_in[10];
  const float* ln1g = (const float*)d_in[11];
  const float* ln1b = (const float*)d_in[12];
  const float* W1 = (const float*)d_in[13];
  const float* b1 = (const float*)d_in[14];
  const float* W2 = (const float*)d_in[15];
  const float* b2 = (const float*)d_in[16];
  const float* ln2g = (const float*)d_in[17];
  const float* ln2b = (const float*)d_in[18];
  float* out = (float*)d_out;

  char* ws = (char*)d_ws;
  const size_t MB = 1u << 20;
  _Float16* xb   = (_Float16*)(ws + 0);        // x_f16, later ctx_f16
  _Float16* wq16 = (_Float16*)(ws + 16 * MB);  // swizzled
  _Float16* wk16 = (_Float16*)(ws + 18 * MB);
  _Float16* wv16 = (_Float16*)(ws + 20 * MB);
  _Float16* wo16 = (_Float16*)(ws + 22 * MB);
  _Float16* w116 = (_Float16*)(ws + 24 * MB);
  _Float16* w216 = (_Float16*)(ws + 26 * MB);
  _Float16* q16  = (_Float16*)(ws + 28 * MB);  // later inner_f16
  _Float16* k16  = (_Float16*)(ws + 44 * MB);  // later h_f16
  _Float16* v16  = (_Float16*)(ws + 60 * MB);
  float*    f32a = (float*)(ws + 76 * MB);     // attn_out, later ffn out2
  float*    hf32 = (float*)(ws + 108 * MB);

  const int NX = 2 * S_LEN * DM_DIM;  // 8388608

  // staging: activations elementwise, weights fragment-swizzled
  cvt_f32_f16<<<(NX + 255) / 256, 256, 0, stream>>>(x, xb, NX);
  cvt_w_swizzle<<<256, 256, 0, stream>>>(Wq, wq16);
  cvt_w_swizzle<<<256, 256, 0, stream>>>(Wk, wk16);
  cvt_w_swizzle<<<256, 256, 0, stream>>>(Wv, wv16);
  cvt_w_swizzle<<<256, 256, 0, stream>>>(Wo, wo16);
  cvt_w_swizzle<<<256, 256, 0, stream>>>(W1, w116);
  cvt_w_swizzle<<<256, 256, 0, stream>>>(W2, w216);

  const dim3 gg(DM_DIM / 256, (2 * S_LEN) / 128);  // (4, 64)
  const dim3 gb(256);

  // QKV projections (bias fused, f16 outputs)
  gemm_wmma_f16<false, false><<<gg, gb, 0, stream>>>(
      xb, wq16, bq, nullptr, q16, 2 * S_LEN, DM_DIM, DM_DIM);
  gemm_wmma_f16<false, false><<<gg, gb, 0, stream>>>(
      xb, wk16, bk, nullptr, k16, 2 * S_LEN, DM_DIM, DM_DIM);
  gemm_wmma_f16<false, false><<<gg, gb, 0, stream>>>(
      xb, wv16, bv, nullptr, v16, 2 * S_LEN, DM_DIM, DM_DIM);

  // BigBird attention -> ctx f16 (reuses xb region; x f16 no longer needed)
  bigbird_attn<<<2 * H_NUM * M_BLKS, 256, 0, stream>>>(q16, k16, v16, mask,
                                                       randb, xb);

  // Output projection -> f32
  gemm_wmma_f16<false, true><<<gg, gb, 0, stream>>>(
      xb, wo16, bo, f32a, nullptr, 2 * S_LEN, DM_DIM, DM_DIM);

  // h = LN(attn + x): f32 + f16 (f16 into k16 region)
  ln_res_kernel<true><<<2 * S_LEN, 256, 0, stream>>>(f32a, x, ln1g, ln1b,
                                                     hf32, k16);

  // FFN: inner = relu(h*W1 + b1) f16 (into q16 region)
  gemm_wmma_f16<true, false><<<gg, gb, 0, stream>>>(
      k16, w116, b1, nullptr, q16, 2 * S_LEN, DM_DIM, DM_DIM);
  // out2 = inner*W2 + b2  f32 (reuse f32a)
  gemm_wmma_f16<false, true><<<gg, gb, 0, stream>>>(
      q16, w216, b2, f32a, nullptr, 2 * S_LEN, DM_DIM, DM_DIM);

  // final: LN(out2 + h) -> d_out (f32)
  ln_res_kernel<false><<<2 * S_LEN, 256, 0, stream>>>(f32a, hf32, ln2g, ln2b,
                                                      out, nullptr);
}
